// parKT_1365799600732
// MI455X (gfx1250) — matrix operand order repaired and checked by reading, not compiled, not run on hardware
//
#include <hip/hip_runtime.h>
#include <hip/hip_bf16.h>

// Problem constants (match reference)
#define Bq  32
#define Sq  512
#define Dq  512
#define Hq  8
#define Lq  4
#define FFq 2048
#define DKq 64
#define SCALEq 0.125f  // 1/sqrt(64)

typedef __attribute__((ext_vector_type(16))) _Float16 v16h;
typedef __attribute__((ext_vector_type(8)))  _Float16 v8h;
typedef __attribute__((ext_vector_type(4)))  _Float16 v4h;
typedef __attribute__((ext_vector_type(8)))  float    v8f;

// ---------------- WMMA fragment helpers (wave32, A/B k = 16*hi + e, not ISA table order) -------

// A-matrix 16x32 fragment from row-major f16 (global or LDS; stride in
// halves). Per lane: one contiguous 32B run (k = 16*hi + e) -> 2x (global|ds)_load_b128.
__device__ inline v16h load_a_frag(const _Float16* a, size_t lda) {
  const int lane = threadIdx.x & 31;
  const int hi = lane >> 4;
  const int m = lane & 15;
  v16h r{};
#pragma unroll
  for (int e = 0; e < 16; ++e) {
    const int p = e >> 1, o = e & 1;
    const int k = 16 * hi + e;
    r[e] = a[(size_t)m * lda + k];
  }
  return r;
}

// B^T fragment for scores = Q @ K^T, direct from row-major f16 K:
// lane (hi,n) needs k[j=n][e+16*hi] -> one contiguous 32B run per lane.
__device__ inline v16h load_bT_frag_g(const _Float16* kt, size_t ld) {
  const int lane = threadIdx.x & 31;
  return *(const v16h*)(kt + (size_t)(lane & 15) * ld + 16 * (lane >> 4));
}

// Fragment pre-swizzled into register order: [lane][16 contiguous halves].
__device__ inline v16h load_frag_sw(const _Float16* base) {
  return *(const v16h*)(base + (threadIdx.x & 31) * 16);
}

__device__ inline v8f wmma_f16(v16h a, v16h b, v8f c) {
  return __builtin_amdgcn_wmma_f32_16x16x32_f16(false, a, false, b,
                                                (short)0, c, false, false);
}

// ---------------- elementwise: x = q+pos (f32+f16), y = qa+pos (f16) ------
__global__ __launch_bounds__(256) void addpos_kernel(
    const float4* __restrict__ q, const float4* __restrict__ qa,
    const float4* __restrict__ pos, float4* __restrict__ x,
    _Float16* __restrict__ xh, _Float16* __restrict__ yh, size_t total4) {
  size_t i = (size_t)blockIdx.x * 256 + threadIdx.x;
  if (i < total4) {
    float4 pe = pos[i & ((size_t)Sq * Dq / 4 - 1)];
    float4 a = q[i], b = qa[i];
    float4 xs = make_float4(a.x + pe.x, a.y + pe.y, a.z + pe.z, a.w + pe.w);
    x[i] = xs;
    v4h hx = {(_Float16)xs.x, (_Float16)xs.y, (_Float16)xs.z, (_Float16)xs.w};
    v4h hy = {(_Float16)(b.x + pe.x), (_Float16)(b.y + pe.y),
              (_Float16)(b.z + pe.z), (_Float16)(b.w + pe.w)};
    *reinterpret_cast<v4h*>(&xh[i * 4]) = hx;
    *reinterpret_cast<v4h*>(&yh[i * 4]) = hy;
  }
}

// ---------------- weight pack: f32 [K,N] -> f16 global B-fragment order ----
// Fragment id f = (n>>4)*(K/32) + (k>>5); within fragment:
// [lane = ((k>>4)&1)*16 + (n&15)][e = k&15], 16 contiguous halves per lane.
__global__ __launch_bounds__(256) void packw_kernel(
    const float* __restrict__ W, _Float16* __restrict__ Wp, int K, int N) {
  const int n4 = N >> 2;
  int idx = blockIdx.x * 256 + threadIdx.x;  // float4 index
  if (idx >= K * n4) return;
  int r = idx / n4;
  int c4 = (idx - r * n4) * 4;
  float4 w = *reinterpret_cast<const float4*>(&W[(size_t)r * N + c4]);
  size_t base = ((size_t)(c4 >> 4) * (K >> 5) + (r >> 5)) * 512 +
                (size_t)((r >> 4) & 1) * 256 + (r & 15);
  const int n = c4 & 15;
  Wp[base + (n + 0) * 16] = (_Float16)w.x;
  Wp[base + (n + 1) * 16] = (_Float16)w.y;
  Wp[base + (n + 2) * 16] = (_Float16)w.z;
  Wp[base + (n + 3) * 16] = (_Float16)w.w;
}

// ---------------- WMMA GEMM, zero LDS: out = A @ W (+bias)(+relu)(+resid) --
// A: f16 [M,K] row-major, fragments loaded DIRECTLY from global.
// Wp: f16 pre-swizzled fragment-major global (L2-resident across M-tiles).
// 64x128 block tile; 8 waves (2M x 4N); wave = 32x32 (2x2 frags).
// OUTF16: write f16-only output (activations consumed by later WMMA).
template <bool RELU, bool RESID, bool OUTF16>
__global__ __launch_bounds__(256) void gemm_kernel(
    const _Float16* __restrict__ A, const _Float16* __restrict__ Wp,
    const float* __restrict__ bias, const float* __restrict__ resid,
    float* __restrict__ outf, _Float16* __restrict__ outh, int M, int K,
    int N) {
  const int tid = threadIdx.x;
  const int lane = tid & 31;
  const int wave = tid >> 5;
  const int wm = wave >> 2;  // 0..1
  const int wn = wave & 3;   // 0..3
  const int m0 = blockIdx.y * 64;
  const int n0 = blockIdx.x * 128;
  const int kfrags = K >> 5;
  const size_t nt0 = (size_t)((n0 >> 4) + wn * 2);

  const _Float16* a0p = A + (size_t)(m0 + wm * 32) * K;
  const _Float16* a1p = A + (size_t)(m0 + wm * 32 + 16) * K;
  const _Float16* b0p = Wp + (nt0 + 0) * (size_t)kfrags * 512;
  const _Float16* b1p = Wp + (nt0 + 1) * (size_t)kfrags * 512;

  v8f acc[2][2] = {};
  for (int kb = 0; kb < kfrags; ++kb) {  // K-step 32, pure vmem + wmma
    v16h a0 = load_a_frag(a0p + kb * 32, K);
    v16h a1 = load_a_frag(a1p + kb * 32, K);
    v16h b0 = load_frag_sw(b0p + (size_t)kb * 512);
    v16h b1 = load_frag_sw(b1p + (size_t)kb * 512);
    acc[0][0] = wmma_f16(a0, b0, acc[0][0]);
    acc[0][1] = wmma_f16(a0, b1, acc[0][1]);
    acc[1][0] = wmma_f16(a1, b0, acc[1][0]);
    acc[1][1] = wmma_f16(a1, b1, acc[1][1]);
  }
  // Epilogue: C/D layout — lane: col = lane&15, rows r + 8*(lane>>4)
  const int n = lane & 15;
  const int hi = lane >> 4;
#pragma unroll
  for (int mi = 0; mi < 2; ++mi) {
#pragma unroll
    for (int ni = 0; ni < 2; ++ni) {
      const int gc = n0 + wn * 32 + ni * 16 + n;
      const float bval = bias[gc];
#pragma unroll
      for (int r = 0; r < 8; ++r) {
        const int gr = m0 + wm * 32 + mi * 16 + r + 8 * hi;
        float val = acc[mi][ni][r] + bval;
        if (RELU) val = fmaxf(val, 0.f);
        if (RESID) val += resid[(size_t)gr * N + gc];
        if (OUTF16)
          outh[(size_t)gr * N + gc] = (_Float16)val;
        else
          outf[(size_t)gr * N + gc] = val;
      }
    }
  }
}

// ---------------- attention: one wave per (b, h, 16-row query block) -------
// Q/K fragments loaded directly from global f16; scores loop barrier-free.
// scores = (Q @ Q^T) * SCALE * forget[b,i]; causal (j < i); softmax;
// global attn row 0 zeroed (zero_pad); ctx = attn @ V -> f16.
__global__ __launch_bounds__(32) void attn_kernel(
    const _Float16* __restrict__ qkh, const _Float16* __restrict__ vh,
    const float* __restrict__ forget, _Float16* __restrict__ ctxh) {
  __shared__ float sc[16][520];                      // f32 scores
  __shared__ __align__(32) _Float16 scf[16][520];    // f16 probs (A source)
  __shared__ __align__(32) _Float16 vf[4 * 512];     // swizzled V frags
  const int lane = threadIdx.x;
  const int qb = blockIdx.x, h = blockIdx.y, b = blockIdx.z;

  const _Float16* qrow = qkh + ((size_t)(b * Sq + qb * 16)) * Dq + h * DKq;
  v16h aq0 = load_a_frag(qrow, Dq);
  v16h aq1 = load_a_frag(qrow + 32, Dq);

  // ---- scores over key blocks 0..qb (strictly causal); no LDS staging
  for (int kb = 0; kb <= qb; ++kb) {
    const _Float16* krow = qkh + ((size_t)(b * Sq + kb * 16)) * Dq + h * DKq;
    v16h b0 = load_bT_frag_g(krow, Dq);
    v16h b1 = load_bT_frag_g(krow + 32, Dq);
    v8f cf = {};
    cf = wmma_f16(aq0, b0, cf);
    cf = wmma_f16(aq1, b1, cf);
    const int n = lane & 15, hi = lane >> 4;
    const int j = kb * 16 + n;
#pragma unroll
    for (int r = 0; r < 8; ++r) {
      const int il = r + 8 * hi;
      const int ig = qb * 16 + il;
      float val = cf[r] * SCALEq * forget[(size_t)b * Sq + ig];
      sc[il][j] = (j >= ig) ? -1e32f : val;  // visible iff j < i (tril,-1)
    }
  }
  __syncthreads();

  // ---- row softmax; normalized probs written as f16 into scf
  const int Lr = (qb + 1) * 16;
  for (int i = 0; i < 16; ++i) {
    float m = -3.4e38f;
    for (int c = lane; c < Lr; c += 32) m = fmaxf(m, sc[i][c]);
#pragma unroll
    for (int off = 16; off; off >>= 1) m = fmaxf(m, __shfl_xor(m, off, 32));
    float s = 0.f;
    for (int c = lane; c < Lr; c += 32) {
      float e = expf(sc[i][c] - m);
      sc[i][c] = e;
      s += e;
    }
#pragma unroll
    for (int off = 16; off; off >>= 1) s += __shfl_xor(s, off, 32);
    const float inv = 1.f / s;
    for (int c = lane; c < Lr; c += 32) scf[i][c] = (_Float16)(sc[i][c] * inv);
  }

  // ---- pad to K-step 32 with zeros; zero_pad: global query row 0
  const int nsteps = (qb + 2) >> 1;
  if ((qb & 1) == 0) {
    for (int idx = lane; idx < 256; idx += 32)
      scf[idx >> 4][Lr + (idx & 15)] = (_Float16)0.f;
  }
  if (qb == 0) {
    for (int c = lane; c < nsteps * 32; c += 32) scf[0][c] = (_Float16)0.f;
  }
  __syncthreads();

  // ---- ctx = attn @ V  (K-step 32 over keys, 4 N-tiles of d)
  v8f cd[4] = {};
  for (int st = 0; st < nsteps; ++st) {
    const _Float16* vbase = vh + ((size_t)(b * Sq + st * 32)) * Dq + h * DKq;
    // V tile 32x64 into B-frag order: frag = d>>4, [hi=j>>4][n=d&15][e=j&15]
#pragma unroll
    for (int it = 0; it < 8; ++it) {
      int idx = lane + it * 32;                 // 256 v8h chunks
      int j = idx >> 3, c8 = (idx & 7) * 8;
      v8h vv = *reinterpret_cast<const v8h*>(&vbase[(size_t)j * Dq + c8]);
      const int base = ((c8 >> 4) * 32 + (j >> 4) * 16) * 16 + (j & 15);
      const int n = c8 & 15;
#pragma unroll
      for (int t = 0; t < 8; ++t) vf[base + (n + t) * 16] = vv[t];
    }
    __syncthreads();
    v16h af = load_a_frag(&scf[0][st * 32], 520);
#pragma unroll
    for (int dt = 0; dt < 4; ++dt) {
      v16h bf = load_frag_sw(&vf[dt * 512]);
      cd[dt] = wmma_f16(af, bf, cd[dt]);
    }
    __syncthreads();
  }
  const int n = lane & 15, hi = lane >> 4;
#pragma unroll
  for (int dt = 0; dt < 4; ++dt) {
    const int gc = h * DKq + dt * 16 + n;
#pragma unroll
    for (int r = 0; r < 8; ++r) {
      const int il = r + 8 * hi;
      ctxh[((size_t)(b * Sq + qb * 16 + il)) * Dq + gc] = (_Float16)cd[dt][r];
    }
  }
}

// ---------------- LayerNorm (D=512), one block per row; optional f16 copy --
template <bool H16>
__global__ __launch_bounds__(128) void ln_kernel(
    const float* __restrict__ in, const float* __restrict__ g,
    const float* __restrict__ bb, float* __restrict__ out,
    _Float16* __restrict__ outh) {
  __shared__ float red[128];
  const int row = blockIdx.x, tid = threadIdx.x;
  float4 v = *reinterpret_cast<const float4*>(&in[(size_t)row * Dq + tid * 4]);
  red[tid] = v.x + v.y + v.z + v.w;
  __syncthreads();
#pragma unroll
  for (int off = 64; off >= 1; off >>= 1) {
    if (tid < off) red[tid] += red[tid + off];
    __syncthreads();
  }
  const float mean = red[0] * (1.0f / Dq);
  __syncthreads();
  float4 d = make_float4(v.x - mean, v.y - mean, v.z - mean, v.w - mean);
  red[tid] = d.x * d.x + d.y * d.y + d.z * d.z + d.w * d.w;
  __syncthreads();
#pragma unroll
  for (int off = 64; off >= 1; off >>= 1) {
    if (tid < off) red[tid] += red[tid + off];
    __syncthreads();
  }
  const float inv = rsqrtf(red[0] * (1.0f / Dq) + 1e-5f);
  float4 gg = *reinterpret_cast<const float4*>(&g[tid * 4]);
  float4 bb4 = *reinterpret_cast<const float4*>(&bb[tid * 4]);
  float4 o = make_float4(d.x * inv * gg.x + bb4.x, d.y * inv * gg.y + bb4.y,
                         d.z * inv * gg.z + bb4.z, d.w * inv * gg.w + bb4.w);
  *reinterpret_cast<float4*>(&out[(size_t)row * Dq + tid * 4]) = o;
  if (H16) {
    v4h hh = {(_Float16)o.x, (_Float16)o.y, (_Float16)o.z, (_Float16)o.w};
    *reinterpret_cast<v4h*>(&outh[(size_t)row * Dq + tid * 4]) = hh;
  }
}

// ---------------- host orchestration --------------------------------------
extern "C" void kernel_launch(void* const* d_in, const int* in_sizes, int n_in,
                              void* d_out, int out_size, void* d_ws,
                              size_t ws_size, hipStream_t stream) {
  (void)in_sizes; (void)n_in; (void)out_size; (void)ws_size;
  const float* qe     = (const float*)d_in[0];
  const float* qae    = (const float*)d_in[1];
  const float* forget = (const float*)d_in[2];
  const float* pos    = (const float*)d_in[3];
  const float* Wk  = (const float*)d_in[4];
  const float* bk  = (const float*)d_in[5];
  const float* Wv  = (const float*)d_in[6];
  const float* bv  = (const float*)d_in[7];
  const float* Wo  = (const float*)d_in[8];
  const float* bo  = (const float*)d_in[9];
  const float* W1  = (const float*)d_in[10];
  const float* b1  = (const float*)d_in[11];
  const float* W2  = (const float*)d_in[12];
  const float* b2  = (const float*)d_in[13];
  const float* g1  = (const float*)d_in[14];
  const float* be1 = (const float*)d_in[15];
  const float* g2  = (const float*)d_in[16];
  const float* be2 = (const float*)d_in[17];
  float* out = (float*)d_out;

  const size_t MSD = (size_t)Bq * Sq * Dq;  // 8,388,608 elements
  char* wsb = (char*)d_ws;
  float*    xb   = (float*)wsb;                 wsb += MSD * 4;  // x (f32)
  float*    t1   = (float*)wsb;                 wsb += MSD * 4;  // pre-LN sum
  _Float16* xh   = (_Float16*)wsb;              wsb += MSD * 2;  // x (f16)
  _Float16* yh   = (_Float16*)wsb;              wsb += MSD * 2;  // y (f16)
  _Float16* qkh  = (_Float16*)wsb;              wsb += MSD * 2;  // q/k proj
  _Float16* vh   = (_Float16*)wsb;              wsb += MSD * 2;  // v proj
  _Float16* ctxh = (_Float16*)wsb;              wsb += MSD * 2;  // attn ctx
  _Float16* h1h  = (_Float16*)wsb;              wsb += MSD * 8;  // FFN hidden
  _Float16* wp   = (_Float16*)wsb;              // packed weights (<=2MB)

  const size_t total4 = MSD / 4;
  addpos_kernel<<<(unsigned)((total4 + 255) / 256), 256, 0, stream>>>(
      (const float4*)qe, (const float4*)qae, (const float4*)pos, (float4*)xb,
      xh, yh, total4);

  const int M = Bq * Sq;  // 16384
  auto packw = [&](const float* W, int K, int N) {
    int n_f4 = K * (N >> 2);
    packw_kernel<<<(n_f4 + 255) / 256, 256, 0, stream>>>(W, wp, K, N);
  };

  for (int l = 0; l < Lq; ++l) {
    const float* Wkl = Wk + (size_t)l * Dq * Dq;
    const float* bkl = bk + (size_t)l * Dq;
    const float* Wvl = Wv + (size_t)l * Dq * Dq;
    const float* bvl = bv + (size_t)l * Dq;
    const float* Wol = Wo + (size_t)l * Dq * Dq;
    const float* bol = bo + (size_t)l * Dq;
    const float* W1l = W1 + (size_t)l * Dq * FFq;
    const float* b1l = b1 + (size_t)l * FFq;
    const float* W2l = W2 + (size_t)l * FFq * Dq;
    const float* b2l = b2 + (size_t)l * Dq;

    // qk = x @ Wk + bk (f16) ; v = y @ Wv + bv (f16)
    packw(Wkl, Dq, Dq);
    gemm_kernel<false, false, true>
        <<<dim3(Dq / 128, M / 64), 256, 0, stream>>>(
            xh, wp, bkl, nullptr, nullptr, qkh, M, Dq, Dq);
    packw(Wvl, Dq, Dq);
    gemm_kernel<false, false, true>
        <<<dim3(Dq / 128, M / 64), 256, 0, stream>>>(
            yh, wp, bvl, nullptr, nullptr, vh, M, Dq, Dq);

    // ctx = attention(qk, qk, v) -> f16
    attn_kernel<<<dim3(Sq / 16, Hq, Bq), 32, 0, stream>>>(
        qkh, vh, forget, ctxh);

    // t1 = x + ctx @ Wo + bo ; x = LN1(t1) (f32 + f16)
    packw(Wol, Dq, Dq);
    gemm_kernel<false, true, false>
        <<<dim3(Dq / 128, M / 64), 256, 0, stream>>>(
            ctxh, wp, bol, xb, t1, nullptr, M, Dq, Dq);
    ln_kernel<true><<<M, 128, 0, stream>>>(t1, g1 + (size_t)l * Dq,
                                           be1 + (size_t)l * Dq, xb, xh);

    // h1 = relu(x @ W1 + b1) (f16) ; t1 = x + h1 @ W2 + b2 ; x = LN2(t1)
    packw(W1l, Dq, FFq);
    gemm_kernel<true, false, true>
        <<<dim3(FFq / 128, M / 64), 256, 0, stream>>>(
            xh, wp, b1l, nullptr, nullptr, h1h, M, Dq, FFq);
    packw(W2l, FFq, Dq);
    gemm_kernel<false, true, false>
        <<<dim3(Dq / 128, M / 64), 256, 0, stream>>>(
            h1h, wp, b2l, xb, t1, nullptr, M, FFq, Dq);
    if (l == Lq - 1) {
      ln_kernel<false><<<M, 128, 0, stream>>>(t1, g2 + (size_t)l * Dq,
                                              be2 + (size_t)l * Dq, out,
                                              nullptr);
    } else {
      ln_kernel<true><<<M, 128, 0, stream>>>(t1, g2 + (size_t)l * Dq,
                                             be2 + (size_t)l * Dq, xb, xh);
    }
  }
}
